// Encoder_52630529245475
// MI455X (gfx1250) — matrix operand order repaired and checked by reading, not compiled, not run on hardware
//
#include <hip/hip_runtime.h>

typedef __attribute__((ext_vector_type(16))) _Float16 v16h;
typedef __attribute__((ext_vector_type(8)))  float    v8f;
typedef __attribute__((ext_vector_type(8)))  unsigned v8u;
typedef __attribute__((ext_vector_type(2)))  unsigned v2u;

#define ENC_UNITS 50
#define FEAT      4
#define TSTEPS    100
#define BATCH     256
#define NGATE     200   // 4*ENC_UNITS
#define NPAD      208   // 13 N-tiles of 16 == 4 * 52 (gate-interleaved, 2 pad units)
#define UPAD      52
#define KPAD      64
#define MTILE     16
#define NWAVES    13
#define BLOCK     (NWAVES * 32)   // 416 threads = 13 waves (wave32)

__device__ __forceinline__ float fast_rcp(float x)     { return __builtin_amdgcn_rcpf(x); }
__device__ __forceinline__ float fast_sigmoid(float x) { return fast_rcp(1.0f + __expf(-x)); }
__device__ __forceinline__ float fast_tanh(float x)    { return __builtin_fmaf(2.0f, fast_sigmoid(x + x), -1.0f); }

__global__ __launch_bounds__(BLOCK, 1) void lstm_idm_kernel(
    const float* __restrict__ x_scaled,     // (256,100,4)
    const float* __restrict__ state,        // (256,100,4)
    const float* __restrict__ lstm_kernel,  // (4,200)
    const float* __restrict__ rec_kernel,   // (50,200)
    const float* __restrict__ lstm_bias,    // (200,)
    const float* __restrict__ w_dv, const float* __restrict__ b_dv,
    const float* __restrict__ w_tg, const float* __restrict__ b_tg,
    const float* __restrict__ w_mj, const float* __restrict__ b_mj,
    const float* __restrict__ w_ma, const float* __restrict__ b_ma,
    const float* __restrict__ w_mi, const float* __restrict__ b_mi,
    float* __restrict__ out)                // [25600 act_seq | 1280 idm_param]
{
    __shared__ __align__(16) _Float16 x16[MTILE][TSTEPS][FEAT];   // 12.8 KB (f16 inputs)
    __shared__ __align__(16) float    z_lds[MTILE][NPAD];         // 13.3 KB
    __shared__ __align__(16) _Float16 h16[2][MTILE][KPAD];        //  4.0 KB (ping-pong)
    __shared__ __align__(16) float    h32[MTILE][UPAD];           //  3.3 KB
    __shared__ float    param_lds[MTILE][8];

    const int tid  = threadIdx.x;
    const int lane = tid & 31;
    const int wave = tid >> 5;     // 0..12 -> N-tile
    const int half = lane >> 4;    // 0/1
    const int nn   = lane & 15;
    const int m0   = blockIdx.x * MTILE;
    const int n0   = wave * 16;

    // Gate-interleaved column permutation: n' = 4*u + g  <->  original col u + 50*g
    const int  ncol   = n0 + nn;
    const int  uu     = ncol >> 2;              // unit 0..51
    const int  gg     = ncol & 3;               // gate 0:i 1:f 2:g 3:o
    const bool uvalid = (uu < ENC_UNITS);
    const int  ocol   = uu + ENC_UNITS * gg;    // column in original (.,200) layout

    // ---------- one-time preload ----------
    for (int i = tid; i < MTILE * TSTEPS * FEAT; i += BLOCK)
        ((_Float16*)x16)[i] = (_Float16)x_scaled[(size_t)m0 * TSTEPS * FEAT + i];
    for (int i = tid; i < 2 * MTILE * KPAD; i += BLOCK)
        ((_Float16*)h16)[i] = (_Float16)0.0f;
    for (int i = tid; i < MTILE * UPAD; i += BLOCK)
        ((float*)h32)[i] = 0.0f;

    // ---------- resident B fragments (permuted columns, zero padded) ----------
    // 16-bit B (32x16), contiguous K order used here (same as A): lanes 0-15 hold K=0..15, lanes 16-31 hold K=16..31;
    // VGPR j holds K = half*16 + 2j, 2j+1 for column N = lane&15.
    // bk: input projection -- rows K=0..3 = lstm_kernel, row K=4 = bias, rest 0.
    v16h b0, b1, bk;
    #pragma unroll
    for (int j = 0; j < 8; ++j) {
        #pragma unroll
        for (int s = 0; s < 2; ++s) {
            int k  = half * 16 + 2 * j + s;
            int k2 = 32 + k;
            b0[2*j+s] = (uvalid && k  < ENC_UNITS) ? (_Float16)rec_kernel[k  * NGATE + ocol] : (_Float16)0.0f;
            b1[2*j+s] = (uvalid && k2 < ENC_UNITS) ? (_Float16)rec_kernel[k2 * NGATE + ocol] : (_Float16)0.0f;
            float kv = 0.0f;
            if (uvalid) {
                if      (k <  FEAT) kv = lstm_kernel[k * NGATE + ocol];
                else if (k == FEAT) kv = lstm_bias[ocol];
            }
            bk[2*j+s] = (_Float16)kv;
        }
    }

    float c0 = 0.0f, c1 = 0.0f;
    const unsigned msel = (half == 0) ? 0xFFFFFFFFu : 0u;  // x/1.0 only live in K=0..15 half
    __syncthreads();

    // ---------- one LSTM step: read hc, write hn; ONE barrier per step ----------
    auto step = [&](const _Float16 (*hc)[KPAD], _Float16 (*hn)[KPAD], int t) {
        // x A-fragment: M=nn row holds K=0..3 = x_t, K=4 = 1.0 (bias lane), rest 0
        v2u xp = *(const v2u*)&x16[nn][t][0];
        v8u axu = {};
        axu[0] = xp[0] & msel;
        axu[1] = xp[1] & msel;
        axu[2] = 0x00003C00u & msel;   // (K=4)=1.0h, (K=5)=0
        v16h ax = __builtin_bit_cast(v16h, axu);

        // h A-fragments (16x32 f16): even K-pairs -> b32/b128 LDS loads
        v8u ap0, ap1;
        const _Float16* hrow = hc[nn];
        #pragma unroll
        for (int j = 0; j < 8; ++j) {
            int kb = half * 16 + 2 * j;  // even
            ap0[j] = *(const unsigned*)&hrow[kb];
            ap1[j] = *(const unsigned*)&hrow[32 + kb];
        }
        v16h a0 = __builtin_bit_cast(v16h, ap0);
        v16h a1 = __builtin_bit_cast(v16h, ap1);

        v8f cacc = {};
        cacc = __builtin_amdgcn_wmma_f32_16x16x32_f16(false, ax, false, bk,
                                                      (short)0, cacc, false, false);
        cacc = __builtin_amdgcn_wmma_f32_16x16x32_f16(false, a0, false, b0,
                                                      (short)0, cacc, false, false);
        cacc = __builtin_amdgcn_wmma_f32_16x16x32_f16(false, a1, false, b1,
                                                      (short)0, cacc, false, false);

        // z tile stays wave-local: store fragment, re-read transposed (in-order DS)
        #pragma unroll
        for (int r = 0; r < 8; ++r)
            z_lds[r + 8 * half][n0 + nn] = cacc[r];
        asm volatile("" ::: "memory");   // keep store->load order; HW LDS is in-order per wave

        // gate update: this wave owns units 4w..4w+3, all 16 rows (2 elems/lane)
        #pragma unroll
        for (int p = 0; p < 2; ++p) {
            int uloc = half + 2 * p;
            float4 zv = *(const float4*)&z_lds[nn][n0 + 4 * uloc]; // (i,f,g,o)
            float ig = fast_sigmoid(zv.x);
            float fg = fast_sigmoid(zv.y);
            float gv = fast_tanh(zv.z);
            float og = fast_sigmoid(zv.w);
            float& cs = (p == 0) ? c0 : c1;
            cs = __builtin_fmaf(fg, cs, ig * gv);
            float h = og * fast_tanh(cs);
            int u = 4 * wave + uloc;        // u<=51; pad units produce h==0
            hn[nn][u]  = (_Float16)h;
            h32[nn][u] = h;
        }
        __syncthreads();   // hn complete before it is read as hc next step
    };

    for (int t = 0; t < TSTEPS; t += 2) {
        step(h16[0], h16[1], t);
        step(h16[1], h16[0], t + 1);
    }

    // ---------- heads: 16 rows x 5 heads ----------
    if (tid < MTILE * 5) {
        int m = tid & 15;
        int j = tid >> 4;
        const float* wv; float bb;
        switch (j) {
            case 0:  wv = w_dv; bb = b_dv[0]; break;
            case 1:  wv = w_tg; bb = b_tg[0]; break;
            case 2:  wv = w_mj; bb = b_mj[0]; break;
            case 3:  wv = w_ma; bb = b_ma[0]; break;
            default: wv = w_mi; bb = b_mi[0]; break;
        }
        float acc = bb;
        #pragma unroll 10
        for (int u = 0; u < ENC_UNITS; ++u) acc += h32[m][u] * wv[u];
        float th = fast_tanh(acc);
        float val;
        if      (j == 0) val = th * 10.00f + 25.00f;   // desired_v  (15,35)
        else if (j == 1) val = th *  1.25f +  1.75f;   // tgap       (0.5,3)
        else if (j == 2) val = fmaxf(acc, 0.0f);       // min_jamx   relu
        else if (j == 3) val = th *  1.25f +  1.75f;   // max_act    (0.5,3)
        else             val = th *  1.75f +  2.25f;   // min_act    (0.5,4)
        param_lds[m][j] = val;
        out[(size_t)BATCH * TSTEPS + (size_t)(m0 + m) * 5 + j] = val;
    }
    __syncthreads();

    // ---------- IDM physics: 16 rows x 100 timesteps ----------
    for (int idx = tid; idx < MTILE * TSTEPS; idx += BLOCK) {
        int m = idx / TSTEPS, t = idx % TSTEPS;
        const float* st = state + ((size_t)(m0 + m) * TSTEPS + t) * FEAT;
        float vel = st[0], dv = st[2], dx = st[3];
        float pdv = param_lds[m][0], ptg = param_lds[m][1], pmj = param_lds[m][2];
        float pma = param_lds[m][3], pmi = param_lds[m][4];
        float two_sqrt_ab = 2.0f * sqrtf(pma * pmi);
        float des_gap = pmj + ptg * vel + vel * dv / two_sqrt_ab;
        float r1 = vel / pdv;  float r1sq = r1 * r1;
        float r2 = des_gap / dx;
        out[(size_t)(m0 + m) * TSTEPS + t] = pma * (1.0f - (r1sq * r1sq + r2 * r2));
    }
}

extern "C" void kernel_launch(void* const* d_in, const int* in_sizes, int n_in,
                              void* d_out, int out_size, void* d_ws, size_t ws_size,
                              hipStream_t stream) {
    (void)in_sizes; (void)n_in; (void)d_ws; (void)ws_size; (void)out_size;
    lstm_idm_kernel<<<dim3(BATCH / MTILE), dim3(BLOCK), 0, stream>>>(
        (const float*)d_in[0],  (const float*)d_in[1],  (const float*)d_in[2],
        (const float*)d_in[3],  (const float*)d_in[4],
        (const float*)d_in[5],  (const float*)d_in[6],
        (const float*)d_in[7],  (const float*)d_in[8],
        (const float*)d_in[9],  (const float*)d_in[10],
        (const float*)d_in[11], (const float*)d_in[12],
        (const float*)d_in[13], (const float*)d_in[14],
        (float*)d_out);
}
